// MixGCN_40441412059141
// MI455X (gfx1250) — compile-verified, matrix operand order checked by reading, not run on hardware
//
#include <hip/hip_runtime.h>
#include <hip/hip_bf16.h>
#include <stddef.h>

// ---------------------------------------------------------------------------
// MixGCN for MI455X (gfx1250): CSR-based GCN aggregation + bf16 WMMA GEMMs.
// ---------------------------------------------------------------------------

#define HIDF 64
#define MOTIF 16
#define TOPK_K 10

typedef __attribute__((ext_vector_type(16))) __bf16 v16bf;
typedef __attribute__((ext_vector_type(8)))  float  v8f;

static __device__ __forceinline__ __bf16 f2bf(float f) {
  unsigned u = __float_as_uint(f);
  u += 0x7FFFu + ((u >> 16) & 1u);           // round-to-nearest-even
  unsigned short hs = (unsigned short)(u >> 16);
  __bf16 r;
  __builtin_memcpy(&r, &hs, sizeof(r));
  return r;
}

// ------------------------------- utility -----------------------------------

__global__ void k_zero_u32(unsigned* p, int n) {
  int i = blockIdx.x * blockDim.x + threadIdx.x;
  if (i < n) p[i] = 0u;
}

__global__ void k_fill_f32(float* p, float v, int n) {
  int i = blockIdx.x * blockDim.x + threadIdx.x;
  if (i < n) p[i] = v;
}

// --------------------------- motif edge weights -----------------------------

__global__ void k_motif_norm(const float* __restrict__ mx, float* __restrict__ mn, int n) {
  int v = blockIdx.x * blockDim.x + threadIdx.x;
  if (v >= n) return;
  float t[MOTIF]; float ss = 0.f;
#pragma unroll
  for (int i = 0; i < MOTIF; ++i) { t[i] = mx[(size_t)v * MOTIF + i]; ss += t[i] * t[i]; }
  float inv = 1.0f / (sqrtf(ss) + 1e-8f);
#pragma unroll
  for (int i = 0; i < MOTIF; ++i) mn[(size_t)v * MOTIF + i] = t[i] * inv;
}

__global__ void k_edge_s(const int* __restrict__ ei, const float* __restrict__ mn,
                         float* __restrict__ s, unsigned* __restrict__ cnt_src, int e_) {
  int e = blockIdx.x * blockDim.x + threadIdx.x;
  if (e >= e_) return;
  int u = ei[e];
  int v = ei[e_ + e];
  float dot = 0.f;
#pragma unroll
  for (int i = 0; i < MOTIF; ++i) dot += mn[(size_t)u * MOTIF + i] * mn[(size_t)v * MOTIF + i];
  s[e] = (dot + 1.0f) * 0.5f;
  atomicAdd(&cnt_src[u], 1u);
}

// ------------------------- single-block exclusive scan ----------------------

__global__ __launch_bounds__(1024) void k_scan(const unsigned* __restrict__ cnt,
                                               unsigned* __restrict__ off, int n) {
  __shared__ unsigned sh[1024];
  __shared__ unsigned carry_s;
  if (threadIdx.x == 0) carry_s = 0u;
  __syncthreads();
  for (int base = 0; base < n; base += 1024) {
    int i = base + (int)threadIdx.x;
    unsigned v = (i < n) ? cnt[i] : 0u;
    sh[threadIdx.x] = v;
    __syncthreads();
#pragma unroll
    for (int d = 1; d < 1024; d <<= 1) {
      unsigned t = (threadIdx.x >= (unsigned)d) ? sh[threadIdx.x - d] : 0u;
      __syncthreads();
      sh[threadIdx.x] += t;
      __syncthreads();
    }
    unsigned incl = sh[threadIdx.x];
    if (i < n) off[i] = carry_s + (incl - v);
    __syncthreads();
    if (threadIdx.x == 1023) carry_s += sh[1023];
    __syncthreads();
  }
  if (threadIdx.x == 0) off[n] = carry_s;
}

// ------------------------------ CSR scatter ---------------------------------

__global__ void k_scatter_src(const int* __restrict__ ei, const float* __restrict__ s,
                              const unsigned* __restrict__ off, unsigned* __restrict__ cur,
                              unsigned* __restrict__ eid, float* __restrict__ ssort, int e_) {
  int e = blockIdx.x * blockDim.x + threadIdx.x;
  if (e >= e_) return;
  int u = ei[e];
  unsigned pos = off[u] + atomicAdd(&cur[u], 1u);
  eid[pos] = (unsigned)e;
  ssort[pos] = s[e];
}

// top-K per source segment: keep iff rank (by s desc, edge-id asc) < K
__global__ void k_topk(const unsigned* __restrict__ off, const unsigned* __restrict__ eid,
                       const float* __restrict__ ssort, float* __restrict__ ew, int n) {
  int v = blockIdx.x * blockDim.x + threadIdx.x;
  if (v >= n) return;
  unsigned a = off[v], b = off[v + 1];
  for (unsigned p = a; p < b; ++p) {
    unsigned ep = eid[p];
    float sp = ssort[p];
    int rank = 0;
    for (unsigned q = a; q < b; ++q) {
      if (q == p) continue;
      float sq = ssort[q];
      unsigned eq = eid[q];
      if (sq > sp || (sq == sp && eq < ep)) ++rank;
    }
    float sc = fminf(fmaxf(sp, 0.0f), 1.0f);
    ew[ep] = (rank < TOPK_K) ? (0.75f + 0.25f * sc) : 0.75f;
  }
}

__global__ void k_deg_hist(const int* __restrict__ ei, const float* __restrict__ ew,
                           float* __restrict__ deg, unsigned* __restrict__ cnt_dst, int e_) {
  int e = blockIdx.x * blockDim.x + threadIdx.x;
  if (e >= e_) return;
  int v = ei[e_ + e];
  atomicAdd(&deg[v], ew[e]);
  atomicAdd(&cnt_dst[v], 1u);
}

__global__ void k_dinv(float* deg, int n) {
  int v = blockIdx.x * blockDim.x + threadIdx.x;
  if (v >= n) return;
  float d = deg[v];                    // >= 1 (self loop), matches reference rsqrt path
  deg[v] = (d > 0.f) ? rsqrtf(fmaxf(d, 1e-30f)) : 0.f;
}

__global__ void k_scatter_dst(const int* __restrict__ ei, const float* __restrict__ ew,
                              const float* __restrict__ dinv, const unsigned* __restrict__ off,
                              unsigned* __restrict__ cur, unsigned* __restrict__ srcs,
                              float* __restrict__ coef, int e_) {
  int e = blockIdx.x * blockDim.x + threadIdx.x;
  if (e >= e_) return;
  int u = ei[e];
  int v = ei[e_ + e];
  unsigned pos = off[v] + atomicAdd(&cur[v], 1u);
  srcs[pos] = (unsigned)u;
  coef[pos] = ew[e] * dinv[u];          // dinv[row] * w ; dinv[col] applied at gather
}

// ---------------------------- WMMA GEMM (bf16) ------------------------------
// OUT[nrows x NVALID] = X[nrows x KD] @ W[KD x NVALID] (+bias).
// One wave per 16-row tile; NTILES 16-col accumulators; K stepped by 32.
template <int KD, int NTILES, int NVALID, bool BIAS>
__global__ __launch_bounds__(128) void k_gemm_wmma(const float* __restrict__ X,
                                                   const float* __restrict__ W,
                                                   const float* __restrict__ bias,
                                                   float* __restrict__ OUT, int nrows) {
  const int lane = threadIdx.x & 31;
  const int tile = blockIdx.x * 4 + (threadIdx.x >> 5);
  const int rowBase = tile * 16;
  if (rowBase >= nrows) return;                 // wave-uniform: EXEC stays all-ones
  const int m = lane & 15;
  const int half = lane >> 4;

  v8f acc[NTILES] = {};
  const float* xrow = X + (size_t)(rowBase + m) * KD;

#pragma unroll
  for (int kc = 0; kc < KD; kc += 32) {
    // A fragment: 16x32 bf16; element i <-> K = kc + (i>>3)*16 + half*8 + (i&7)
    v16bf a;
#pragma unroll
    for (int i = 0; i < 16; ++i) {
      int k = kc + ((i >> 3) << 4) + (half << 3) + (i & 7);
      a[i] = f2bf(xrow[k]);
    }
#pragma unroll
    for (int t = 0; t < NTILES; ++t) {
      const int n = t * 16 + m;
      const bool okn = (n < NVALID);
      v16bf b;
#pragma unroll
      for (int i = 0; i < 16; ++i) {
        int k = kc + ((i >> 3) << 4) + (half << 3) + (i & 7);
        b[i] = okn ? f2bf(W[(size_t)k * NVALID + n]) : f2bf(0.0f);
      }
      acc[t] = __builtin_amdgcn_wmma_f32_16x16x32_bf16(
          false, a, false, b, (short)0, acc[t], false, false);
    }
  }

#pragma unroll
  for (int t = 0; t < NTILES; ++t) {
    const int n = t * 16 + m;
    if (n >= NVALID) continue;
    const float bv = BIAS ? bias[n] : 0.0f;
#pragma unroll
    for (int r = 0; r < 8; ++r) {
      const int row = rowBase + r + (half << 3);   // C/D: M = r + 8*half, N = lane&15
      OUT[(size_t)row * NVALID + n] = acc[t][r] + bv;
    }
  }
}

// -------------------------- GCN gather-aggregation --------------------------
// out[v,f] = dinv[v]*( dinv[v]*h[v,f] + sum_in coef*h[u,f] ) + bias[f]
__global__ __launch_bounds__(64) void k_aggregate(const float* __restrict__ h,
                                                  const unsigned* __restrict__ off,
                                                  const unsigned* __restrict__ srcs,
                                                  const float* __restrict__ coef,
                                                  const float* __restrict__ dinv,
                                                  const float* __restrict__ bias,
                                                  float* __restrict__ out, int n) {
  const int v = blockIdx.x;
  if (v >= n) return;
  const int f = threadIdx.x;                      // 0..63
  const float dv = dinv[v];
  float acc = dv * h[(size_t)v * HIDF + f];       // self loop contribution (w=1)
  const unsigned a = off[v], b = off[v + 1];
  for (unsigned p = a; p < b; ++p) {
    unsigned u = srcs[p];
    if (p + 1 < b)
      __builtin_prefetch(h + (size_t)srcs[p + 1] * HIDF, 0, 1);  // global_prefetch_b8
    acc += coef[p] * h[(size_t)u * HIDF + f];
  }
  out[(size_t)v * HIDF + f] = dv * acc + bias[f];
}

// --------------------- LayerNorm + ReLU + residual add ----------------------
// one wave per row, 2 features per lane (wave32)
__global__ __launch_bounds__(256) void k_ln_relu_add(const float* __restrict__ agg,
                                                     const float* __restrict__ g,
                                                     const float* __restrict__ be,
                                                     const float* __restrict__ res,
                                                     float* __restrict__ out, int n) {
  const int row = blockIdx.x * (blockDim.x >> 5) + (threadIdx.x >> 5);
  if (row >= n) return;
  const int lane = threadIdx.x & 31;
  const float v0 = agg[(size_t)row * HIDF + lane];
  const float v1 = agg[(size_t)row * HIDF + 32 + lane];
  float sum = v0 + v1;
#pragma unroll
  for (int o = 16; o > 0; o >>= 1) sum += __shfl_xor(sum, o, 32);
  const float mu = sum * (1.0f / 64.0f);
  const float d0 = v0 - mu, d1 = v1 - mu;
  float vs = d0 * d0 + d1 * d1;
#pragma unroll
  for (int o = 16; o > 0; o >>= 1) vs += __shfl_xor(vs, o, 32);
  const float inv = rsqrtf(vs * (1.0f / 64.0f) + 1e-5f);
  const float y0 = fmaxf(d0 * inv * g[lane] + be[lane], 0.0f);
  const float y1 = fmaxf(d1 * inv * g[32 + lane] + be[32 + lane], 0.0f);
  out[(size_t)row * HIDF + lane]      = y0 + res[(size_t)row * HIDF + lane];
  out[(size_t)row * HIDF + 32 + lane] = y1 + res[(size_t)row * HIDF + 32 + lane];
}

// ---------------------------------------------------------------------------

extern "C" void kernel_launch(void* const* d_in, const int* in_sizes, int n_in,
                              void* d_out, int out_size, void* d_ws, size_t ws_size,
                              hipStream_t stream) {
  const float* x       = (const float*)d_in[0];
  const float* motif_x = (const float*)d_in[1];
  const int*   ei      = (const int*)d_in[2];
  const float* W0      = (const float*)d_in[3];
  const float* b0      = (const float*)d_in[4];
  const float* W1      = (const float*)d_in[5];
  const float* b1      = (const float*)d_in[6];
  const float* ln0_g   = (const float*)d_in[7];
  const float* ln0_b   = (const float*)d_in[8];
  const float* ln1_g   = (const float*)d_in[9];
  const float* ln1_b   = (const float*)d_in[10];
  const float* res0_W  = (const float*)d_in[11];
  const float* res0_b  = (const float*)d_in[12];
  const float* head_W  = (const float*)d_in[13];
  const float* head_b  = (const float*)d_in[14];
  float* out = (float*)d_out;

  const int n = in_sizes[0] / 128;   // 50000
  const int e = in_sizes[2] / 2;     // 800000

  // ---- workspace carve (all buffers fully written before read) ----
  size_t off = 0;
  auto carve = [&](size_t bytes) -> void* {
    void* p = (char*)d_ws + off;
    off += (bytes + 255) & ~(size_t)255;
    return p;
  };
  float*    mnorm   = (float*)carve((size_t)n * MOTIF * 4);
  float*    s       = (float*)carve((size_t)e * 4);
  float*    ew      = (float*)carve((size_t)e * 4);
  unsigned* cnt     = (unsigned*)carve((size_t)n * 4);
  unsigned* cur     = (unsigned*)carve((size_t)n * 4);
  unsigned* off_src = (unsigned*)carve((size_t)(n + 1) * 4);
  unsigned* off_dst = (unsigned*)carve((size_t)(n + 1) * 4);
  unsigned* eid     = (unsigned*)carve((size_t)e * 4);   // later reused as srcs_sorted
  float*    ssort   = (float*)carve((size_t)e * 4);      // later reused as coef
  float*    deg     = (float*)carve((size_t)n * 4);      // becomes dinv in place
  float*    bufH    = (float*)carve((size_t)n * HIDF * 4);
  float*    bufRes  = (float*)carve((size_t)n * HIDF * 4);
  float*    bufAgg  = (float*)carve((size_t)n * HIDF * 4);
  float*    bufX1   = (float*)carve((size_t)n * HIDF * 4);
  (void)ws_size; (void)n_in; (void)out_size;

  const int TB = 256;
  const int gN = (n + TB - 1) / TB;
  const int gE = (e + TB - 1) / TB;

  // ---- edge weights: cosine -> per-src top-K ----
  k_zero_u32<<<gN, TB, 0, stream>>>(cnt, n);
  k_motif_norm<<<gN, TB, 0, stream>>>(motif_x, mnorm, n);
  k_edge_s<<<gE, TB, 0, stream>>>(ei, mnorm, s, cnt, e);
  k_scan<<<1, 1024, 0, stream>>>(cnt, off_src, n);
  k_zero_u32<<<gN, TB, 0, stream>>>(cur, n);
  k_scatter_src<<<gE, TB, 0, stream>>>(ei, s, off_src, cur, eid, ssort, e);
  k_topk<<<gN, TB, 0, stream>>>(off_src, eid, ssort, ew, n);

  // ---- degree normalization + dst-CSR ----
  k_fill_f32<<<gN, TB, 0, stream>>>(deg, 1.0f, n);       // self-loop weight
  k_zero_u32<<<gN, TB, 0, stream>>>(cnt, n);
  k_deg_hist<<<gE, TB, 0, stream>>>(ei, ew, deg, cnt, e);
  k_scan<<<1, 1024, 0, stream>>>(cnt, off_dst, n);
  k_dinv<<<gN, TB, 0, stream>>>(deg, n);                 // deg -> dinv
  k_zero_u32<<<gN, TB, 0, stream>>>(cur, n);
  k_scatter_dst<<<gE, TB, 0, stream>>>(ei, ew, deg, off_dst, cur, eid, ssort, e);
  unsigned* srcs = eid;   // reused
  float*    coef = ssort; // reused

  const int tiles  = (n + 15) / 16;
  const int gGemm  = (tiles + 3) / 4;  // 4 waves per block

  // ---- layer 0 ----
  k_gemm_wmma<128, 4, HIDF, false><<<gGemm, 128, 0, stream>>>(x, W0, nullptr, bufH, n);
  k_gemm_wmma<128, 4, HIDF, true ><<<gGemm, 128, 0, stream>>>(x, res0_W, res0_b, bufRes, n);
  k_aggregate<<<n, 64, 0, stream>>>(bufH, off_dst, srcs, coef, deg, b0, bufAgg, n);
  k_ln_relu_add<<<(n + 7) / 8, 256, 0, stream>>>(bufAgg, ln0_g, ln0_b, bufRes, bufX1, n);

  // ---- layer 1 (identity residual) ----
  k_gemm_wmma<64, 4, HIDF, false><<<gGemm, 128, 0, stream>>>(bufX1, W1, nullptr, bufH, n);
  k_aggregate<<<n, 64, 0, stream>>>(bufH, off_dst, srcs, coef, deg, b1, bufAgg, n);
  k_ln_relu_add<<<(n + 7) / 8, 256, 0, stream>>>(bufAgg, ln1_g, ln1_b, bufX1, bufRes, n);

  // ---- head: [N,64] @ [64,40] + b -> d_out ----
  k_gemm_wmma<64, 3, 40, true><<<gGemm, 128, 0, stream>>>(bufRes, head_W, head_b, out, n);
}
